// QuantizedLinear_19799799234864
// MI455X (gfx1250) — matrix operand order repaired and checked by reading, not compiled, not run on hardware
//
#include <hip/hip_runtime.h>
#include <cstdint>
#include <cstddef>

// ---------- types ----------
typedef __attribute__((ext_vector_type(16))) __bf16 v16bf;
typedef __attribute__((ext_vector_type(8)))  __bf16 v8bf;
typedef __attribute__((ext_vector_type(4)))  __bf16 v4bf;
typedef __attribute__((ext_vector_type(8)))  float  v8f;
typedef __attribute__((ext_vector_type(4)))  int    v4i;

// pointer-to-vector types in explicit address spaces for the async builtin
typedef __attribute__((address_space(1))) v4i* as1_v4i_ptr;  // global
typedef __attribute__((address_space(3))) v4i* as3_v4i_ptr;  // LDS

// ---------- problem dims (from reference) ----------
constexpr int M = 8192;
constexpr int N = 4096;
constexpr int K = 4096;

// ---------- GEMM tiling ----------
constexpr int BM   = 256;      // block tile M (8 waves as 4x2, wave tile 64x64)
constexpr int BN   = 128;      // block tile N
constexpr int BK   = 64;       // K tile (2 WMMA k-steps)
constexpr int LDSS = BK + 8;   // padded LDS row stride (72 bf16 = 144B, 16B-aligned)

// gfx1250 async global->LDS path, guarded so a missing builtin falls back cleanly
#if defined(__HIP_DEVICE_COMPILE__) && \
    __has_builtin(__builtin_amdgcn_global_load_async_to_lds_b128) && \
    __has_builtin(__builtin_amdgcn_s_wait_asynccnt)
#define USE_ASYNC_LDS 1
#else
#define USE_ASYNC_LDS 0
#endif

__device__ __forceinline__ void copy16_g2l(const __bf16* g, __bf16* l) {
#if USE_ASYNC_LDS
  // 16B per lane, tracked by ASYNCcnt; LDS offset = low 32 bits of generic ptr
  __builtin_amdgcn_global_load_async_to_lds_b128(
      (as1_v4i_ptr)(uintptr_t)g,
      (as3_v4i_ptr)(uint32_t)(uintptr_t)l,
      0, 0);
#else
  const uint4 v = *(const uint4*)g;
  *(uint4*)l = v;
#endif
}

__device__ __forceinline__ v16bf frag_cat(v8bf lo, v8bf hi) {
  return __builtin_shufflevector(lo, hi,
      0, 1, 2, 3, 4, 5, 6, 7, 8, 9, 10, 11, 12, 13, 14, 15);
}

// ---------- prepass 1: group-128 int4 quant-dequant of W -> bf16 ----------
// one wave (32 lanes) per group; 4 contiguous floats per lane; shuffle-reduce absmax
__global__ __launch_bounds__(256) void quant_w_bf16(const float* __restrict__ w,
                                                    __bf16* __restrict__ wq) {
  const int group = (blockIdx.x * 256 + threadIdx.x) >> 5;
  const int lane  = threadIdx.x & 31;
  const size_t base = (size_t)group * 128 + (size_t)lane * 4;

  const float4 v = *(const float4*)(w + base);
  float m = fmaxf(fmaxf(fabsf(v.x), fabsf(v.y)), fmaxf(fabsf(v.z), fabsf(v.w)));
#pragma unroll
  for (int off = 16; off > 0; off >>= 1)
    m = fmaxf(m, __shfl_xor(m, off, 32));

  float scale = m * (1.0f / 7.0f);
  scale = (scale > 0.0f) ? scale : 1.0f;
  const float inv = 1.0f / scale;

  const float q0 = fminf(fmaxf(rintf(v.x * inv), -8.0f), 7.0f);
  const float q1 = fminf(fmaxf(rintf(v.y * inv), -8.0f), 7.0f);
  const float q2 = fminf(fmaxf(rintf(v.z * inv), -8.0f), 7.0f);
  const float q3 = fminf(fmaxf(rintf(v.w * inv), -8.0f), 7.0f);

  v4bf o = { (__bf16)(q0 * scale), (__bf16)(q1 * scale),
             (__bf16)(q2 * scale), (__bf16)(q3 * scale) };
  *(v4bf*)(wq + base) = o;
}

// ---------- prepass 2: x f32 -> bf16 ----------
__global__ __launch_bounds__(256) void cvt_x_bf16(const float* __restrict__ x,
                                                  __bf16* __restrict__ xb) {
  const size_t i = ((size_t)blockIdx.x * 256 + threadIdx.x) * 4;
  const float4 v = *(const float4*)(x + i);
  v4bf o = { (__bf16)v.x, (__bf16)v.y, (__bf16)v.z, (__bf16)v.w };
  *(v4bf*)(xb + i) = o;
}

// ---------- main GEMM: y = xb @ wq^T + bias ----------
// 256 threads = 8 waves in a 4(M)x2(N) grid; each wave owns a 64x64 output tile
// = 16x v_wmma_f32_16x16x32_bf16 per 32-wide k-step. Double-buffered LDS with
// async global->LDS copies overlapped with compute (ASYNCcnt pipeline).
__global__ __launch_bounds__(256) void gemm_bf16_wmma(const __bf16* __restrict__ xb,
                                                      const __bf16* __restrict__ wq,
                                                      const float*  __restrict__ bias,
                                                      float* __restrict__ out) {
  constexpr int ABUF = BM * LDSS;       // 256*72 = 18432 elems
  constexpr int BBUF = BN * LDSS;       // 128*72 =  9216 elems
  constexpr int BUF  = ABUF + BBUF;     // 27648 elems (55296 B); x2 buffers = 108 KB
  __shared__ __align__(16) __bf16 smem[2 * BUF];

  const int tid  = threadIdx.x;
  const int lane = tid & 31;
  const int wave = tid >> 5;
  const int wm   = wave >> 1;   // 0..3  (M direction)
  const int wn   = wave & 1;    // 0..1  (N direction)
  const int lr   = lane & 15;   // row within 16x16 fragment
  const int lh   = lane >> 4;   // half-wave selector

  const int nblk = blockIdx.x * BN;
  const int mblk = blockIdx.y * BM;

  // global->LDS copy assignment:
  //   A: 1 thread per row (256 rows), 64 bf16 = 8x16B chunks
  //   B: 2 threads per row (128 rows), 32 bf16 = 4x16B chunks each
  const int arow = tid;
  const int brow = tid >> 1;
  const int bcol = (tid & 1) * 32;
  const __bf16* gA = xb + (size_t)(mblk + arow) * K;
  const __bf16* gB = wq + (size_t)(nblk + brow) * K + bcol;

  auto copy_tile = [&](int kt, int buf) {
    __bf16* base = smem + buf * BUF;
    __bf16* lA = base + arow * LDSS;
    __bf16* lB = base + ABUF + brow * LDSS + bcol;
#pragma unroll
    for (int i = 0; i < 8; ++i)
      copy16_g2l(gA + kt + i * 8, lA + i * 8);
#pragma unroll
    for (int i = 0; i < 4; ++i)
      copy16_g2l(gB + kt + i * 8, lB + i * 8);
  };

  v8f acc[4][4] = {};

  copy_tile(0, 0);                         // prologue: stage first tile
  for (int kt = 0; kt < K; kt += BK) {
    const int cur = (kt / BK) & 1;
#if USE_ASYNC_LDS
    __builtin_amdgcn_s_wait_asynccnt(0);   // my copies into buf[cur] landed
#endif
    __syncthreads();                       // everyone's landed; prev compute done
    if (kt + BK < K)
      copy_tile(kt + BK, cur ^ 1);         // async prefetch next tile during compute

    const __bf16* Abase = smem + cur * BUF;
    const __bf16* Bbase = Abase + ABUF;
#pragma unroll
    for (int kb = 0; kb < BK; kb += 32) {
      v16bf a[4];
      v16bf b[4];
      // A fragment (16x32): lanes<16 hold K {0..7,16..23}, lanes>=16 {8..15,24..31}
#pragma unroll
      for (int mt = 0; mt < 4; ++mt) {
        const __bf16* p = Abase + (wm * 64 + mt * 16 + lr) * LDSS + kb + lh * 8;
        a[mt] = frag_cat(*(const v8bf*)p, *(const v8bf*)(p + 16));
      }
      // B fragment (32x16): lanes<16 hold K {0..7,16..23}, lanes>=16 {8..15,24..31}
#pragma unroll
      for (int nt = 0; nt < 4; ++nt) {
        const __bf16* p = Bbase + (wn * 64 + nt * 16 + lr) * LDSS + kb + lh * 8;
        b[nt] = frag_cat(*(const v8bf*)p, *(const v8bf*)(p + 16));
      }
#pragma unroll
      for (int mt = 0; mt < 4; ++mt)
#pragma unroll
        for (int nt = 0; nt < 4; ++nt)
          acc[mt][nt] = __builtin_amdgcn_wmma_f32_16x16x32_bf16(
              false, a[mt], false, b[nt], (short)0, acc[mt][nt], false, false);
    }
  }

  // epilogue: C/D layout VGPR i -> row (i + 8*lh), col = lane%16
#pragma unroll
  for (int nt = 0; nt < 4; ++nt) {
    const int col = nblk + wn * 64 + nt * 16 + lr;
    const float bv = bias[col];
#pragma unroll
    for (int mt = 0; mt < 4; ++mt) {
      const int rbase = mblk + wm * 64 + mt * 16 + lh * 8;
#pragma unroll
      for (int i = 0; i < 8; ++i)
        out[(size_t)(rbase + i) * N + col] = acc[mt][nt][i] + bv;
    }
  }
}

extern "C" void kernel_launch(void* const* d_in, const int* in_sizes, int n_in,
                              void* d_out, int out_size, void* d_ws, size_t ws_size,
                              hipStream_t stream) {
  const float* x    = (const float*)d_in[0];   // [8192, 4096]
  const float* w    = (const float*)d_in[1];   // [4096, 4096]
  const float* bias = (const float*)d_in[2];   // [4096]
  float* out = (float*)d_out;                  // [8192, 4096]

  // workspace layout: wq bf16 [N*K] (32MB) | xb bf16 [M*K] (64MB)
  __bf16* wq = (__bf16*)d_ws;
  __bf16* xb = (__bf16*)((char*)d_ws + (size_t)N * K * sizeof(__bf16));

  {
    const int groups = (N * K) / 128;           // 131072 groups, 1 wave each
    quant_w_bf16<<<groups / 8, 256, 0, stream>>>(w, wq);
  }
  {
    const int threads = (M * K) / 4;            // 4 elems per thread
    cvt_x_bf16<<<threads / 256, 256, 0, stream>>>(x, xb);
  }
  {
    dim3 grid(N / BN, M / BM);                  // (32, 32)
    gemm_bf16_wmma<<<grid, 256, 0, stream>>>(xb, wq, bias, out);
  }
}
